// PaiNNInteraction_48206712930390
// MI455X (gfx1250) — matrix operand order repaired and checked by reading, not compiled, not run on hardware
//
#include <hip/hip_runtime.h>
#include <hip/hip_bf16.h>

// PaiNN interaction for MI455X (gfx1250, wave32).
//  K1: seed d_out with (q, mu)                       (outputs are input + segment_sum)
//  K2a: pre-pack W1/W2 -> f16 WMMA B-fragment layout (once, tiny)
//  K2b: x = silu(q@W1+b1)@W2+b2 via v_wmma_f32_16x16x32_f16, x -> d_ws
//  K3: per-edge gather/scale/scatter; Wij streamed with NT hint so x/mu stay L2-resident

typedef __attribute__((ext_vector_type(16))) _Float16 v16h;
typedef __attribute__((ext_vector_type(8)))  float    v8f;
typedef __attribute__((ext_vector_type(4)))  float    f4;

#define F_DIM  128
#define F3_DIM 384
#define NT1 8    // N-tiles in GEMM1 (128/16)
#define NT2 24   // N-tiles in GEMM2 (384/16)

// ---------------------------------------------------------------- K1: init out
__global__ __launch_bounds__(256) void painn_init_out(
    const float4* __restrict__ q, const float4* __restrict__ mu,
    float4* __restrict__ out_q, float4* __restrict__ out_mu,
    int nq4, int nmu4)
{
    int g = blockIdx.x * blockDim.x + threadIdx.x;
    if (g < nq4)  out_q[g]  = q[g];
    if (g < nmu4) out_mu[g] = mu[g];
}

// ---------------------------------------------------------------- K2a: pack weights
// Layout: [ntile][kb][lane][t] halves; per-lane v16h is one contiguous 32B chunk.
// B 32x16 fragment (ISA 7.12.2): K = kb*32 + (t>>3)*16 + (lane>>4)*8 + (t&7) (t=0..15); N = lane&15.
__global__ __launch_bounds__(256) void painn_pack_w(
    const float* __restrict__ W1, const float* __restrict__ W2,
    _Float16* __restrict__ W1p, _Float16* __restrict__ W2p)
{
    int g = blockIdx.x * blockDim.x + threadIdx.x;
    int t    = g & 15;
    int lane = (g >> 4) & 31;
    int kb   = (g >> 9) & 3;
    int nt   = g >> 11;
    int n = nt * 16 + (lane & 15);
    int k = kb * 32 + (t >> 3) * 16 + (lane >> 4) * 8 + (t & 7);
    if (g < NT1 * 4 * 32 * 16) W1p[g] = (_Float16)W1[(size_t)k * F_DIM  + n];
    if (g < NT2 * 4 * 32 * 16) W2p[g] = (_Float16)W2[(size_t)k * F3_DIM + n];
}

// ---------------------------------------------------------------- K2b: MLP (WMMA)
// One block = 16 atoms, 8 waves. GEMM1: wave w -> N-tile w. GEMM2: tiles w, w+8, w+16.
__global__ __launch_bounds__(256) void painn_mlp_wmma(
    const float* __restrict__ q,
    const _Float16* __restrict__ W1p, const float* __restrict__ b1,
    const _Float16* __restrict__ W2p, const float* __restrict__ b2,
    float* __restrict__ x, int n_atoms)
{
    __shared__ _Float16 hlds[16][F_DIM + 8];   // +8 halves pad: conflict-free column reads

    const int a0   = blockIdx.x * 16;
    const int lane = threadIdx.x & 31;
    const int wave = threadIdx.x >> 5;
    const int m    = lane & 15;
    const int hi   = lane >> 4;

    const v16h* W1v = (const v16h*)W1p;        // index: (nt*4 + kb)*32 + lane
    const v16h* W2v = (const v16h*)W2p;

    // ---- A fragments from q[a0+m][0..127]  (ISA 16-bit A 16x32 layout)
    v16h afrag[4];
#pragma unroll
    for (int kb = 0; kb < 4; ++kb) {
#pragma unroll
        for (int t = 0; t < 16; ++t) {
            int k = kb * 32 + (t >> 3) * 16 + hi * 8 + (t & 7);
            afrag[kb][t] = (_Float16)q[(size_t)(a0 + m) * F_DIM + k];
        }
    }

    // ---- GEMM1: h = silu(q@W1 + b1); wave owns columns [wave*16, wave*16+16)
    {
        const int n = wave * 16 + m;
        v8f acc = {};
#pragma unroll
        for (int kb = 0; kb < 4; ++kb) {
            v16h bfrag = W1v[(wave * 4 + kb) * 32 + lane];
            acc = __builtin_amdgcn_wmma_f32_16x16x32_f16(
                false, afrag[kb], false, bfrag, (short)0, acc, false, false);
        }
        const float bias = b1[n];
#pragma unroll
        for (int r = 0; r < 8; ++r) {          // C/D: M = r + 8*hi, N = n
            float v = acc[r] + bias;
            v = v * (1.0f / (1.0f + __expf(-v)));   // silu
            hlds[r + 8 * hi][n] = (_Float16)v;
        }
    }
    __syncthreads();

    // ---- rebuild A fragments from h (LDS)
#pragma unroll
    for (int kb = 0; kb < 4; ++kb) {
#pragma unroll
        for (int t = 0; t < 16; ++t) {
            int k = kb * 32 + (t >> 3) * 16 + hi * 8 + (t & 7);
            afrag[kb][t] = hlds[m][k];
        }
    }

    // ---- GEMM2: x = h@W2 + b2; 24 N-tiles over 8 waves, fully unrolled (12 WMMAs/wave)
#pragma unroll
    for (int it = 0; it < 3; ++it) {
        const int nt = wave + it * 8;
        const int n  = nt * 16 + m;
        v8f acc = {};
#pragma unroll
        for (int kb = 0; kb < 4; ++kb) {
            v16h bfrag = W2v[(nt * 4 + kb) * 32 + lane];
            acc = __builtin_amdgcn_wmma_f32_16x16x32_f16(
                false, afrag[kb], false, bfrag, (short)0, acc, false, false);
        }
        const float bias = b2[n];
#pragma unroll
        for (int r = 0; r < 8; ++r) {
            int row = a0 + r + 8 * hi;
            if (row < n_atoms) x[(size_t)row * F3_DIM + n] = acc[r] + bias;
        }
    }
}

// ---------------------------------------------------------------- K3: edges
// One wave per edge, thread c = 4-feature group. Wij/dir are single-use -> NT loads
// (keeps the x/mu gather working set, 154MB, resident in the 192MB L2).
__global__ __launch_bounds__(256) void painn_edge(
    const float* __restrict__ x,      // [n_atoms, 384]  (d_ws)
    const float* __restrict__ mu,     // [n_atoms, 3, 128]
    const float* __restrict__ Wij,    // [E, 384]
    const float* __restrict__ dir,    // [E, 3]
    const int* __restrict__ idx_i,
    const int* __restrict__ idx_j,
    float* __restrict__ out_q,        // [n_atoms, 128]
    float* __restrict__ out_mu,       // [n_atoms, 3, 128]
    int n_edges)
{
    int gid = blockIdx.x * blockDim.x + threadIdx.x;
    int e  = gid >> 5;                 // wave-uniform edge id
    int c  = gid & 31;                 // float4 group: features [4c, 4c+4)
    if (e >= n_edges) return;

    const int j = idx_j[e];
    const int i = idx_i[e];
    const float d0 = __builtin_nontemporal_load(dir + 3 * (size_t)e + 0);
    const float d1 = __builtin_nontemporal_load(dir + 3 * (size_t)e + 1);
    const float d2 = __builtin_nontemporal_load(dir + 3 * (size_t)e + 2);

    const f4* w  = (const f4*)(Wij + (size_t)e * F3_DIM);
    const f4* xj = (const f4*)(x   + (size_t)j * F3_DIM);
    const f4* mj = (const f4*)(mu  + (size_t)j * F3_DIM);

    const f4 w0 = __builtin_nontemporal_load(w + c);
    const f4 w1 = __builtin_nontemporal_load(w + 32 + c);
    const f4 w2 = __builtin_nontemporal_load(w + 64 + c);
    const f4 x0 = xj[c], x1 = xj[32 + c], x2 = xj[64 + c];
    const f4 m0 = mj[c], m1 = mj[32 + c], m2 = mj[64 + c];

    const f4 dq  = w0 * x0;            // dq contribution
    const f4 dmr = w1 * x1;            // dmuR
    const f4 dmm = w2 * x2;            // dmumu
    const f4 u0 = dmr * d0 + dmm * m0;
    const f4 u1 = dmr * d1 + dmm * m1;
    const f4 u2 = dmr * d2 + dmm * m2;

    float* oq = out_q  + (size_t)i * F_DIM  + 4 * c;
    float* om = out_mu + (size_t)i * F3_DIM + 4 * c;
#pragma unroll
    for (int s = 0; s < 4; ++s) {
        atomicAdd(oq + s,        dq[s]);
        atomicAdd(om + s,        u0[s]);
        atomicAdd(om + 128 + s,  u1[s]);
        atomicAdd(om + 256 + s,  u2[s]);
    }
}

// ---------------------------------------------------------------- launch
extern "C" void kernel_launch(void* const* d_in, const int* in_sizes, int n_in,
                              void* d_out, int out_size, void* d_ws, size_t ws_size,
                              hipStream_t stream)
{
    const float* q   = (const float*)d_in[0];
    const float* mu  = (const float*)d_in[1];
    const float* Wij = (const float*)d_in[2];
    const float* dir = (const float*)d_in[3];
    const float* W1  = (const float*)d_in[4];
    const float* b1  = (const float*)d_in[5];
    const float* W2  = (const float*)d_in[6];
    const float* b2  = (const float*)d_in[7];
    const int* idx_i = (const int*)d_in[8];
    const int* idx_j = (const int*)d_in[9];

    const int n_atoms = in_sizes[0] / F_DIM;   // q is [n_atoms, 1, 128]
    const int n_edges = in_sizes[8];           // idx_i length

    float* out_q  = (float*)d_out;
    float* out_mu = out_q + (size_t)n_atoms * F_DIM;

    // workspace: x [n_atoms*384] f32, then packed f16 weights
    float* x = (float*)d_ws;
    _Float16* W1p = (_Float16*)((char*)d_ws + (size_t)n_atoms * F3_DIM * sizeof(float));
    _Float16* W2p = W1p + NT1 * 4 * 32 * 16;   // 16384 halves, keeps 32B alignment

    // K1: seed outputs with q, mu
    {
        int nq4  = n_atoms * F_DIM  / 4;
        int nmu4 = n_atoms * F3_DIM / 4;
        int n    = (nmu4 > nq4 ? nmu4 : nq4);
        painn_init_out<<<(n + 255) / 256, 256, 0, stream>>>(
            (const float4*)q, (const float4*)mu,
            (float4*)out_q, (float4*)out_mu, nq4, nmu4);
    }

    // K2a: pack weights into WMMA B-fragment layout (f16)
    {
        int total = NT2 * 4 * 32 * 16;         // 49152 >= W1 pack size too
        painn_pack_w<<<(total + 255) / 256, 256, 0, stream>>>(W1, W2, W1p, W2p);
    }

    // K2b: per-atom MLP -> x (WMMA)
    {
        int grid = (n_atoms + 15) / 16;
        painn_mlp_wmma<<<grid, 256, 0, stream>>>(q, W1p, b1, W2p, b2, x, n_atoms);
    }

    // K3: edge gather/scatter
    {
        long total = (long)n_edges * 32;
        int grid = (int)((total + 255) / 256);
        painn_edge<<<grid, 256, 0, stream>>>(x, mu, Wij, dir, idx_i, idx_j,
                                             out_q, out_mu, n_edges);
    }
}
